// EfficientAttention_79508434583873
// MI455X (gfx1250) — matrix operand order repaired and checked by reading, not compiled, not run on hardware
//
#include <hip/hip_runtime.h>

// ---------------------------------------------------------------------------
// EfficientAttention for MI455X (gfx1250): fused QKV GEMM + RoPE + flash
// attention + output projection.
//  - matrix math:      v_wmma_f32_16x16x32_f16 (f32 accumulate)
//  - GEMM staging:     TDM tensor_load_to_lds (TENSORcnt), double-buffered LDS
//  - attention staging: global_load_async_to_lds_b128 (ASYNCcnt)
// B=16, S=512, H=2048, NH=16, NKV=4, HD=128.
// ---------------------------------------------------------------------------

typedef __attribute__((ext_vector_type(16))) _Float16     v16h;
typedef __attribute__((ext_vector_type(8)))  _Float16     v8h;
typedef __attribute__((ext_vector_type(8)))  float        v8f;
typedef __attribute__((ext_vector_type(4)))  unsigned int v4u;
typedef __attribute__((ext_vector_type(8)))  unsigned int v8u;

#define B_  16
#define S_  512
#define H_  2048
#define NH_ 16
#define NKV_ 4
#define HD_ 128
#define LDQKV 3072   // q(2048) | k(512) | v(512)

// Low 32 bits of a generic shared-aperture address == LDS byte offset
// (flat->LDS mapping truncates to addr[31:0]).
__device__ __forceinline__ unsigned int lds_off(const void* p) {
  return (unsigned int)(unsigned long long)p;
}

// gfx1250 async global->LDS copy, 16 bytes per lane, tracked by ASYNCcnt.
__device__ __forceinline__ void async_copy_b128(unsigned int lds_byte_off,
                                                const void* gaddr) {
  asm volatile("global_load_async_to_lds_b128 %0, %1, off"
               :: "v"(lds_byte_off), "v"(gaddr)
               : "memory");
}
__device__ __forceinline__ void wait_async0() {
  asm volatile("s_wait_asynccnt 0" ::: "memory");
}
__device__ __forceinline__ void wait_tensor0() {
  __builtin_amdgcn_s_wait_tensorcnt((short)0);
}

// ---------------------------------------------------------------------------
// TDM: 2D tile load global->LDS via tensor descriptor (D#), 2-group form.
// data_size = 2 bytes (f16). Optional LDS row padding via pad fields.
// Must be issued by a single wave per destination tile (EXEC is ignored,
// but each wave that executes the instruction issues a full transfer).
// ---------------------------------------------------------------------------
__device__ __forceinline__ void tdm_load_2d_f16(
    unsigned int lds_byte,        // LDS byte offset of tile start
    const void* gaddr,            // global address of tile start
    unsigned int tensor_w,        // tensor dim0 length (elements)
    unsigned int tensor_h,        // tensor dim1 length (rows)
    unsigned int tile_w,          // tile dim0 (elements per row)
    unsigned int tile_h,          // tile dim1 (rows)
    unsigned int stride_elems,    // dim0 stride (elements)
    unsigned int pad_enable,      // 1 = pad LDS rows
    unsigned int pad_interval,    // 2^(v+1) DWORDs between pads
    unsigned int pad_amount) {    // (v+1) DWORDs of pad
  unsigned long long ga = (unsigned long long)gaddr;
  v4u g0;
  g0[0] = 1u;                                        // count=1, user mode
  g0[1] = lds_byte;                                  // lds_addr
  g0[2] = (unsigned int)ga;                          // global_addr[31:0]
  g0[3] = (unsigned int)(ga >> 32) | (2u << 30);     // addr[56:32] | type=2
  v8u g1;
  g1[0] = (1u << 16)                                 // data_size: 2 bytes
        | (pad_enable << 20)
        | (pad_interval << 22)
        | (pad_amount << 25);                        // wg_mask=0 (no cluster)
  g1[1] = (tensor_w & 0xFFFFu) << 16;                // tensor_dim0[15:0]
  g1[2] = (tensor_w >> 16) | ((tensor_h & 0xFFFFu) << 16);
  g1[3] = (tensor_h >> 16) | (tile_w << 16);         // tile_dim0
  g1[4] = tile_h;                                    // tile_dim1 (tile_dim2=0)
  g1[5] = stride_elems;                              // tensor_dim0_stride lo32
  g1[6] = 0;                                         // stride hi | dim1_stride
  g1[7] = 0;
  asm volatile("tensor_load_to_lds %0, %1" :: "s"(g0), "s"(g1) : "memory");
}

// ---------------------------------------------------------------------------
// fp32 -> f16 conversion (bandwidth-trivial prep so GEMMs run on WMMA f16)
// ---------------------------------------------------------------------------
__global__ void cvt_f32_to_f16(const float* __restrict__ src,
                               _Float16* __restrict__ dst, int n) {
  int i = (blockIdx.x * blockDim.x + threadIdx.x) * 4;
  if (i < n) {
    float4 v = *(const float4*)(src + i);
    dst[i + 0] = (_Float16)v.x;
    dst[i + 1] = (_Float16)v.y;
    dst[i + 2] = (_Float16)v.z;
    dst[i + 3] = (_Float16)v.w;
  }
}

// ---------------------------------------------------------------------------
// GEMM: C[M,N] = A[M,K] (f16) * W[N,K]^T (f16), f32 accumulate.
// 128x128 tile per 256-thread workgroup (8 waves), K-step 32.
// Double-buffered LDS fed by the Tensor Data Mover; one barrier per K-step.
// Each wave owns a 32x64 sub-tile = 2x4 WMMA accumulators.
// ---------------------------------------------------------------------------
#define GT_M 128
#define GT_N 128
#define GT_K 32
#define LPAD 8   // 8 halves = 4 DWORDs of LDS row padding

template <typename OutT>
__global__ __launch_bounds__(256) void gemm_xwt(
    const _Float16* __restrict__ A, const _Float16* __restrict__ W,
    OutT* __restrict__ C, int M, int N, int K) {
  __shared__ _Float16 sA[2][GT_M][GT_K + LPAD];
  __shared__ _Float16 sB[2][GT_N][GT_K + LPAD];

  const int tid  = threadIdx.x;
  const int lane = tid & 31;
  const int wid  = tid >> 5;
  const int wm   = wid >> 1;        // 0..3  (32-row strip)
  const int wn   = wid & 1;         // 0..1  (64-col strip)
  const int bm   = blockIdx.y;
  const int bn   = blockIdx.x;

  const int m16   = lane & 15;
  const int kb8   = (lane >> 4) << 3;   // unused; A-frag also indexes with ks16
  const int ks16  = (lane >> 4) << 4;   // B-frag: +16 K for upper lane half
  const int rbase = (lane >> 4) << 3;   // C-layout: +8 rows for upper half

  v8f acc[2][4];
#pragma unroll
  for (int i = 0; i < 2; ++i)
#pragma unroll
    for (int j = 0; j < 4; ++j) acc[i][j] = (v8f){};

  const int nsteps = K / GT_K;

  // TDM stage of K-slice s into LDS buffer `buf`: two 128x32 f16 tiles.
  // Rows are 16 DWORDs -> pad_interval code 3; pad 4 DWORDs -> code 3,
  // matching the sA/sB row stride of (32+8) halves.
  auto stage = [&](int s, int buf) {
    if (tid < 32) {  // single wave issues the DMA
      const int k0 = s * GT_K;
      tdm_load_2d_f16(lds_off(&sA[buf][0][0]),
                      A + (size_t)(bm * GT_M) * K + k0,
                      K, GT_M, GT_K, GT_M, K, 1, 3, 3);
      tdm_load_2d_f16(lds_off(&sB[buf][0][0]),
                      W + (size_t)(bn * GT_N) * K + k0,
                      K, GT_N, GT_K, GT_N, K, 1, 3, 3);
    }
  };

  stage(0, 0);
  wait_tensor0();
  __syncthreads();

  int buf = 0;
  for (int s = 0; s < nsteps; ++s) {
    if (s + 1 < nsteps) stage(s + 1, buf ^ 1);  // DMA overlaps WMMAs below

    v16h af[2], bf[4];
#pragma unroll
    for (int i = 0; i < 2; ++i) {
      const _Float16* pa = &sA[buf][wm * 32 + i * 16 + m16][0];
#pragma unroll
      for (int e = 0; e < 8; ++e) {
        af[i][e]     = pa[ks16 + e];
        af[i][8 + e] = pa[ks16 + 8 + e];
      }
    }
#pragma unroll
    for (int j = 0; j < 4; ++j) {
      const _Float16* pb = &sB[buf][wn * 64 + j * 16 + m16][ks16];
#pragma unroll
      for (int e = 0; e < 16; ++e) bf[j][e] = pb[e];
    }
#pragma unroll
    for (int i = 0; i < 2; ++i)
#pragma unroll
      for (int j = 0; j < 4; ++j)
        acc[i][j] = __builtin_amdgcn_wmma_f32_16x16x32_f16(
            false, af[i], false, bf[j], (short)0, acc[i][j], false, false);

    wait_tensor0();    // next slice landed (issued before compute)
    __syncthreads();   // everyone done reading `buf`, DMA data visible
    buf ^= 1;
  }

#pragma unroll
  for (int i = 0; i < 2; ++i)
#pragma unroll
    for (int j = 0; j < 4; ++j)
#pragma unroll
      for (int r = 0; r < 8; ++r) {
        int row = bm * GT_M + wm * 32 + i * 16 + rbase + r;
        int col = bn * GT_N + wn * 64 + j * 16 + m16;
        C[(size_t)row * N + col] = (OutT)acc[i][j][r];
      }
}

// ---------------------------------------------------------------------------
// RoPE applied in place to q (heads 0..15) and k (heads 16..19) of qkv.
// Each thread owns a (d, d+64) pair -> race-free in-place rotation.
// ---------------------------------------------------------------------------
__global__ void rope_kernel(_Float16* __restrict__ qkv) {
  int idx = blockIdx.x * blockDim.x + threadIdx.x;  // < rows*20*64
  int d   = idx & 63;
  int t   = idx >> 6;
  int hh  = t % 20;        // 0..15 q heads, 16..19 k heads (col = hh*128)
  int row = t / 20;        // b*S + s
  int s   = row & (S_ - 1);

  // inv_freq[d] = 10000^(-d/64) = exp(-d * ln(10000)/64)
  float inv = __expf(-(float)d * 0.14391156831f);
  float ang = (float)s * inv;
  float sn, cs;
  __sincosf(ang, &sn, &cs);

  _Float16* p = qkv + (size_t)row * LDQKV + hh * 128 + d;
  float x1 = (float)p[0];
  float x2 = (float)p[64];
  p[0]  = (_Float16)(x1 * cs - x2 * sn);
  p[64] = (_Float16)(x2 * cs + x1 * sn);
}

// ---------------------------------------------------------------------------
// Flash attention: one WG per (b, h, 128 q-rows). 8 waves x 16 q-rows each.
// K-tile staged via async-to-LDS; V^T staged with a manual transpose.
// Online softmax with xor-shuffle row reductions inside 16-lane half-groups.
// ---------------------------------------------------------------------------
__global__ __launch_bounds__(256) void flash_attn(
    const _Float16* __restrict__ qkv, _Float16* __restrict__ out) {
  const int qt  = blockIdx.x;   // 0..3   (128-row q tile)
  const int h   = blockIdx.y;   // 0..15
  const int b   = blockIdx.z;   // 0..15
  const int kvh = h >> 2;       // GQA: 4 q heads share one kv head

  const int tid  = threadIdx.x;
  const int lane = tid & 31;
  const int wid  = tid >> 5;

  const int m16   = lane & 15;
  const int kb8   = (lane >> 4) << 3;
  const int ks16  = (lane >> 4) << 4;
  const int rbase = (lane >> 4) << 3;

  const _Float16* qbase = qkv + (size_t)(b * S_) * LDQKV + h * HD_;
  const _Float16* kbase = qkv + (size_t)(b * S_) * LDQKV + 2048 + kvh * HD_;
  const _Float16* vbase = kbase + 512;

  __shared__ _Float16 sK[32][HD_ + 8];       // [key][d]
  __shared__ _Float16 sVT[HD_][32 + 8];      // [d][key] (transposed stage)
  __shared__ _Float16 sP[8][16][32 + 8];     // per-wave prob tile, C->A relayout

  // staging chunks (key row, 8-half segment)
  const int c0key = tid >> 4,          c0seg = (tid & 15) * 8;
  const int c1key = (tid + 256) >> 4,  c1seg = ((tid + 256) & 15) * 8;

  // Q fragments for this wave's 16 rows, pre-scaled by 1/sqrt(HD)
  const float scale = 0.08838834764831845f;  // 128^-0.5
  v16h qf[4];
  {
    const _Float16* qr = qbase + (size_t)(qt * 128 + wid * 16 + m16) * LDQKV;
#pragma unroll
    for (int c = 0; c < 4; ++c)
#pragma unroll
      for (int e = 0; e < 8; ++e) {
        qf[c][e]     = (_Float16)((float)qr[c * 32 + ks16 + e] * scale);
        qf[c][8 + e] = (_Float16)((float)qr[c * 32 + ks16 + 8 + e] * scale);
      }
  }

  v8f oacc[8];
#pragma unroll
  for (int dt = 0; dt < 8; ++dt) oacc[dt] = (v8f){};
  float mrun[8], lrun[8];
#pragma unroll
  for (int r = 0; r < 8; ++r) { mrun[r] = -3.0e38f; lrun[r] = 0.f; }

  for (int kt = 0; kt < S_ / 32; ++kt) {
    // K tile: async straight into LDS (no VGPR round-trip)
    async_copy_b128(lds_off(&sK[c0key][c0seg]),
                    kbase + (size_t)(kt * 32 + c0key) * LDQKV + c0seg);
    async_copy_b128(lds_off(&sK[c1key][c1seg]),
                    kbase + (size_t)(kt * 32 + c1key) * LDQKV + c1seg);
    // V tile: load + transpose into sVT (async/TDM cannot transpose)
#pragma unroll
    for (int i = 0; i < 2; ++i) {
      int key = (i == 0) ? c0key : c1key;
      int seg = (i == 0) ? c0seg : c1seg;
      v8h vv = *(const v8h*)(vbase + (size_t)(kt * 32 + key) * LDQKV + seg);
#pragma unroll
      for (int e = 0; e < 8; ++e) sVT[seg + e][key] = vv[e];
    }
    wait_async0();
    __syncthreads();

    // scores: Q(16x128) @ K_tile^T(128x32) -> two 16x16 C tiles
    v8f c0 = (v8f){}, c1 = (v8f){};
#pragma unroll
    for (int dc = 0; dc < 4; ++dc) {
      v16h b0, b1;
      const _Float16* p0 = &sK[m16][dc * 32 + ks16];
      const _Float16* p1 = &sK[16 + m16][dc * 32 + ks16];
#pragma unroll
      for (int e = 0; e < 16; ++e) { b0[e] = p0[e]; b1[e] = p1[e]; }
      c0 = __builtin_amdgcn_wmma_f32_16x16x32_f16(false, qf[dc], false, b0,
                                                  (short)0, c0, false, false);
      c1 = __builtin_amdgcn_wmma_f32_16x16x32_f16(false, qf[dc], false, b1,
                                                  (short)0, c1, false, false);
    }

    // online softmax: row reductions across 16-lane half-groups
#pragma unroll
    for (int r = 0; r < 8; ++r) {
      float s0 = c0[r], s1 = c1[r];
      float mx = fmaxf(s0, s1);
#pragma unroll
      for (int off = 1; off < 16; off <<= 1)
        mx = fmaxf(mx, __shfl_xor(mx, off, 32));
      float mo = mrun[r];
      float mn = fmaxf(mo, mx);
      mrun[r]  = mn;
      float alpha = __expf(mo - mn);
      float e0 = __expf(s0 - mn);
      float e1 = __expf(s1 - mn);
      float rs = e0 + e1;
#pragma unroll
      for (int off = 1; off < 16; off <<= 1) rs += __shfl_xor(rs, off, 32);
      lrun[r] = lrun[r] * alpha + rs;
#pragma unroll
      for (int dt = 0; dt < 8; ++dt) oacc[dt][r] *= alpha;
      sP[wid][rbase + r][m16]      = (_Float16)e0;
      sP[wid][rbase + r][16 + m16] = (_Float16)e1;
    }

    // P (C layout) -> A fragment via per-wave LDS region (in-order DS ops)
    v16h pf;
    {
      const _Float16* pp = &sP[wid][m16][0];
#pragma unroll
      for (int e = 0; e < 8; ++e) {
        pf[e]     = pp[ks16 + e];
        pf[8 + e] = pp[ks16 + 8 + e];
      }
    }

    // O += P(16x32) @ V_tile(32x128)
#pragma unroll
    for (int dt = 0; dt < 8; ++dt) {
      v16h vf;
      const _Float16* pv = &sVT[dt * 16 + m16][ks16];
#pragma unroll
      for (int e = 0; e < 16; ++e) vf[e] = pv[e];
      oacc[dt] = __builtin_amdgcn_wmma_f32_16x16x32_f16(
          false, pf, false, vf, (short)0, oacc[dt], false, false);
    }
    __syncthreads();
  }

  // normalize and write [b*S + q][h*128 + d] (f16, feeds output projection)
#pragma unroll
  for (int r = 0; r < 8; ++r) {
    float inv = 1.0f / lrun[r];
    int row   = b * S_ + qt * 128 + wid * 16 + rbase + r;
#pragma unroll
    for (int dt = 0; dt < 8; ++dt)
      out[(size_t)row * H_ + h * HD_ + dt * 16 + m16] =
          (_Float16)(oacc[dt][r] * inv);
  }
}

// ---------------------------------------------------------------------------
// launch
// ---------------------------------------------------------------------------
extern "C" void kernel_launch(void* const* d_in, const int* in_sizes, int n_in,
                              void* d_out, int out_size, void* d_ws,
                              size_t ws_size, hipStream_t stream) {
  const float* x  = (const float*)d_in[0];
  const float* Wq = (const float*)d_in[1];
  const float* Wk = (const float*)d_in[2];
  const float* Wv = (const float*)d_in[3];
  const float* Wo = (const float*)d_in[4];

  const int M = B_ * S_;          // 8192
  const int K = H_;               // 2048
  const int Nqkv = LDQKV;         // 3072
  const int Nkv  = NKV_ * HD_;    // 512

  _Float16* ws     = (_Float16*)d_ws;
  _Float16* xh     = ws;                                  // [8192,2048]
  _Float16* Wcat   = xh + (size_t)M * K;                  // [3072,2048]
  _Float16* Woh    = Wcat + (size_t)Nqkv * K;             // [2048,2048]
  _Float16* qkv    = Woh + (size_t)H_ * K;                // [8192,3072]
  _Float16* attn_o = qkv + (size_t)M * Nqkv;              // [8192,2048]

  // 1) fp32 -> f16 conversions (Wq|Wk|Wv concatenated along N)
  cvt_f32_to_f16<<<(M * K) / 1024, 256, 0, stream>>>(x, xh, M * K);
  cvt_f32_to_f16<<<(H_ * K) / 1024, 256, 0, stream>>>(Wq, Wcat, H_ * K);
  cvt_f32_to_f16<<<(Nkv * K) / 1024, 256, 0, stream>>>(
      Wk, Wcat + (size_t)H_ * K, Nkv * K);
  cvt_f32_to_f16<<<(Nkv * K) / 1024, 256, 0, stream>>>(
      Wv, Wcat + (size_t)(H_ + Nkv) * K, Nkv * K);
  cvt_f32_to_f16<<<(H_ * K) / 1024, 256, 0, stream>>>(Wo, Woh, H_ * K);

  // 2) fused QKV projection: qkv = xh @ Wcat^T
  gemm_xwt<_Float16><<<dim3(Nqkv / GT_N, M / GT_M), 256, 0, stream>>>(
      xh, Wcat, qkv, M, Nqkv, K);

  // 3) RoPE in place on q and k heads
  rope_kernel<<<(M * 20 * 64) / 256, 256, 0, stream>>>(qkv);

  // 4) flash attention
  flash_attn<<<dim3(S_ / 128, NH_, B_), 256, 0, stream>>>(qkv, attn_o);

  // 5) output projection straight to f32 d_out
  gemm_xwt<float><<<dim3(H_ / GT_N, M / GT_M), 256, 0, stream>>>(
      attn_o, Woh, (float*)d_out, M, H_, K);
}
